// ColumnMLPs_59794534695163
// MI455X (gfx1250) — matrix operand order repaired and checked by reading, not compiled, not run on hardware
//
#include <hip/hip_runtime.h>
#include <hip/hip_bf16.h>
#include <math.h>

typedef _Float16 v16h __attribute__((ext_vector_type(16)));
typedef _Float16 v8h  __attribute__((ext_vector_type(8)));
typedef float    v8f  __attribute__((ext_vector_type(8)));

#define D_S 256
#define HDQ 64
#define KK 32            // neighbors
#define MB 32            // rows per node_kernel block

// LDS strides (in halves), padded, 16-byte aligned
#define XS  336          // x tile: 320 cols
#define H1S 1040         // h1 tile: 1024 cols
#define QHS 144          // q-hidden tile: 128 cols
#define IDS 80           // node_id tile: 64 cols

// dynamic-LDS partition for node_kernel (offsets in halves)
#define SM_XT  0
#define SM_H1  (MB * XS)
#define SM_QH  (SM_H1 + MB * H1S)
#define SM_TOTAL (SM_QH + MB * QHS)          // halves

// packed f16 weight offsets (in halves) inside d_ws
#define OFF_CW1 0                       // 320*1024
#define OFF_CW2 (OFF_CW1 + 320*1024)    // 1024*256
#define OFF_QW1 (OFF_CW2 + 1024*256)    // 320*128
#define OFF_QW2 (OFF_QW1 + 320*128)     // 128*64
#define OFF_KW1 (OFF_QW2 + 128*64)      // 64*128
#define OFF_KW2 (OFF_KW1 + 64*128)      // 128*64
#define PACK_TOTAL (OFF_KW2 + 128*64)   // 655360 halves
#define Q_BYTE_OFF ((size_t)PACK_TOTAL * 2)        // f32 q[N*64]

// tanh-GELU via sigmoid identity: branchless, one v_exp_f32 + one divide.
__device__ __forceinline__ float gelu_tanh(float x) {
    float u = x * (1.0f + 0.044715f * x * x);
    return x / (1.0f + __expf(-1.5957691216057308f * u));
}

// Build 16x32 f16 A-fragment from an LDS tile (row-major, `stride` halves).
// Lane l holds row M=l&15. lanes 0-15: K = kbase+[0..15];
// lanes 16-31: K = kbase+[16..31] (matches packed B layout).
__device__ __forceinline__ v16h load_a_frag(const _Float16* lds, int lane,
                                            int kbase, int stride) {
    int row = lane & 15;
    const _Float16* p = lds + row * stride + kbase + ((lane >> 4) << 4);
    union { v16h v; v8h h[2]; } u;
    u.h[0] = *(const v8h*)(p);
    u.h[1] = *(const v8h*)(p + 8);
    return u.v;
}

// B-fragment: packed layout -> one contiguous 32B load per lane
__device__ __forceinline__ v16h load_b_frag(const _Float16* P, int tile_n,
                                            int nslabs, int ks, int lane) {
    return *(const v16h*)(P + (((size_t)tile_n * nslabs + ks) << 9) + lane * 16);
}

__device__ __forceinline__ v8f wmma_f16(v16h a, v16h b, v8f c) {
    return __builtin_amdgcn_wmma_f32_16x16x32_f16(false, a, false, b,
                                                  (short)0, c, false, false);
}

// ---------------- weight packing ----------------
__device__ __forceinline__ void pack_one(const float* W, _Float16* P,
                                         int Kdim, int Ndim, int idx) {
    int k = idx / Ndim, n = idx - k * Ndim;
    int lane = (n & 15) + (((k & 31) >> 4) << 4);
    int e = k & 15;
    size_t off = (((size_t)(n >> 4) * (Kdim >> 5) + (k >> 5)) << 9) + lane * 16 + e;
    P[off] = (_Float16)W[(size_t)k * Ndim + n];
}

__global__ __launch_bounds__(256) void pack_kernel(
    const float* cW1, const float* cW2, const float* qW1,
    const float* qW2, const float* kW1, const float* kW2, _Float16* P) {
    int idx = blockIdx.x * 256 + threadIdx.x;
    const int s1 = 320 * 1024, s2 = s1 + 1024 * 256, s3 = s2 + 320 * 128;
    const int s4 = s3 + 128 * 64, s5 = s4 + 64 * 128, s6 = s5 + 128 * 64;
    if (idx < s1)      pack_one(cW1, P + OFF_CW1, 320, 1024, idx);
    else if (idx < s2) pack_one(cW2, P + OFF_CW2, 1024, 256, idx - s1);
    else if (idx < s3) pack_one(qW1, P + OFF_QW1, 320, 128, idx - s2);
    else if (idx < s4) pack_one(qW2, P + OFF_QW2, 128, 64, idx - s3);
    else if (idx < s5) pack_one(kW1, P + OFF_KW1, 64, 128, idx - s4);
    else if (idx < s6) pack_one(kW2, P + OFF_KW2, 128, 64, idx - s5);
}

// ---------------- k = mlp(node_id) : M=16 blocks ----------------
__global__ __launch_bounds__(128) void k_mlp_kernel(
    const float* __restrict__ node_id,
    const _Float16* __restrict__ PW1, const float* __restrict__ b1,
    const _Float16* __restrict__ PW2, const float* __restrict__ b2,
    float* __restrict__ k_out, int N) {
    __shared__ _Float16 xt[16 * IDS];
    __shared__ _Float16 ht[16 * QHS];
    int tid = threadIdx.x, wave = tid >> 5, lane = tid & 31;
    long long row0 = (long long)blockIdx.x * 16;

    for (int i = tid; i < 16 * 64; i += 128) {
        int r = i >> 6, c = i & 63;
        long long gr = row0 + r; if (gr >= N) gr = N - 1;
        xt[r * IDS + c] = (_Float16)node_id[gr * 64 + c];
    }
    __syncthreads();

    {   // layer1: [16,64]x[64,128], wave owns 2 of 8 col-tiles; K-slabs=2
        v8f acc[2] = {};
        for (int ks = 0; ks < 2; ++ks) {
            v16h a = load_a_frag(xt, lane, ks * 32, IDS);
            #pragma unroll
            for (int t = 0; t < 2; ++t)
                acc[t] = wmma_f16(a, load_b_frag(PW1, wave * 2 + t, 2, ks, lane), acc[t]);
        }
        int rbase = (lane >> 4) << 3;
        #pragma unroll
        for (int t = 0; t < 2; ++t) {
            int col = (wave * 2 + t) * 16 + (lane & 15);
            float bias = b1[col];
            #pragma unroll
            for (int r = 0; r < 8; ++r)
                ht[(rbase + r) * QHS + col] = (_Float16)gelu_tanh(acc[t][r] + bias);
        }
    }
    __syncthreads();

    {   // layer2: [16,128]x[128,64], wave owns 1 of 4 col-tiles; K-slabs=4
        v8f acc = {};
        for (int ks = 0; ks < 4; ++ks)
            acc = wmma_f16(load_a_frag(ht, lane, ks * 32, QHS),
                           load_b_frag(PW2, wave, 4, ks, lane), acc);
        int col = wave * 16 + (lane & 15);
        float bias = b2[col];
        int rbase = (lane >> 4) << 3;
        #pragma unroll
        for (int r = 0; r < 8; ++r) {
            long long gr = row0 + rbase + r;
            if (gr < N) k_out[gr * 64 + col] = acc[r] + bias;
        }
    }
}

// ---------------- m_out + q : M=32 blocks, dynamic LDS ----------------
__global__ __launch_bounds__(256) void node_kernel(
    const float* __restrict__ s, const float* __restrict__ node_id,
    const float* __restrict__ g,
    const _Float16* __restrict__ PcW1, const float* __restrict__ cb1,
    const _Float16* __restrict__ PcW2, const float* __restrict__ cb2,
    const _Float16* __restrict__ PqW1, const float* __restrict__ qb1,
    const _Float16* __restrict__ PqW2, const float* __restrict__ qb2,
    float* __restrict__ m_out, float* __restrict__ q_out, int N) {
    extern __shared__ _Float16 smem[];
    _Float16* xt = smem + SM_XT;   // [MB, XS]
    _Float16* h1 = smem + SM_H1;   // [MB, H1S]
    _Float16* qh = smem + SM_QH;   // [MB, QHS]
    int tid = threadIdx.x, wave = tid >> 5, lane = tid & 31;
    long long row0 = (long long)blockIdx.x * MB;

    // ---- stage 1: RMSNorm(s)*g -> xt[:,0:256]; node_id -> xt[:,256:320]
    {
        int r = tid >> 3;          // row 0..31 (8 threads per row)
        int c8 = tid & 7;          // each thread: 32 contiguous cols
        long long gr = row0 + r; if (gr >= N) gr = N - 1;
        const float* srow = s + gr * 256;
        float vals[32]; float ss = 0.f;
        #pragma unroll
        for (int i = 0; i < 32; ++i) {
            float v = srow[c8 * 32 + i];
            vals[i] = v; ss += v * v;
        }
        // reduce over the 8 lanes of this row (xor stays inside 8-lane group)
        #pragma unroll
        for (int m = 4; m >= 1; m >>= 1) ss += __shfl_xor(ss, m, 32);
        float scale = rsqrtf(ss * (1.0f / 256.0f) + 1e-6f);
        #pragma unroll
        for (int i = 0; i < 32; ++i) {
            int c = c8 * 32 + i;
            xt[r * XS + c] = (_Float16)(vals[i] * scale * g[c]);
        }
        for (int i = tid; i < MB * 64; i += 256) {
            int rr = i >> 6, cc = i & 63;
            long long g2 = row0 + rr; if (g2 >= N) g2 = N - 1;
            xt[rr * XS + 256 + cc] = (_Float16)node_id[g2 * 64 + cc];
        }
    }
    __syncthreads();

    // ---- stage 2: h1 = gelu(x@cW1+cb1) [32,1024]; qh = gelu(x@qW1+qb1) [32,128]
    // K-outer: 2 A-loads (row-halves) feed 18 WMMAs (9 col-tiles x 2 halves).
    {
        v8f acc[9][2] = {};
        for (int ks = 0; ks < 10; ++ks) {
            if (ks + 1 < 10)
                __builtin_prefetch(PcW1 + (((size_t)(wave * 8) * 10 + ks + 1) << 9)
                                        + lane * 16, 0, 1);
            v16h a0 = load_a_frag(xt,           lane, ks * 32, XS);
            v16h a1 = load_a_frag(xt + 16 * XS, lane, ks * 32, XS);
            #pragma unroll
            for (int t = 0; t < 8; ++t) {
                v16h b = load_b_frag(PcW1, wave * 8 + t, 10, ks, lane);
                acc[t][0] = wmma_f16(a0, b, acc[t][0]);
                acc[t][1] = wmma_f16(a1, b, acc[t][1]);
            }
            {
                v16h b = load_b_frag(PqW1, wave, 10, ks, lane);
                acc[8][0] = wmma_f16(a0, b, acc[8][0]);
                acc[8][1] = wmma_f16(a1, b, acc[8][1]);
            }
        }
        int rbase = (lane >> 4) << 3;
        #pragma unroll
        for (int t = 0; t < 8; ++t) {
            int col = (wave * 8 + t) * 16 + (lane & 15);
            float bias = cb1[col];
            #pragma unroll
            for (int h = 0; h < 2; ++h)
                #pragma unroll
                for (int r = 0; r < 8; ++r)
                    h1[(h * 16 + rbase + r) * H1S + col] =
                        (_Float16)gelu_tanh(acc[t][h][r] + bias);
        }
        {
            int col = wave * 16 + (lane & 15);
            float bias = qb1[col];
            #pragma unroll
            for (int h = 0; h < 2; ++h)
                #pragma unroll
                for (int r = 0; r < 8; ++r)
                    qh[(h * 16 + rbase + r) * QHS + col] =
                        (_Float16)gelu_tanh(acc[8][h][r] + bias);
        }
    }
    __syncthreads();

    // ---- stage 3: m_out = h1@cW2+cb2 [32,256]; q = qh@qW2+qb2 [32,64]
    {
        v8f acc[2][2] = {};
        for (int ks = 0; ks < 32; ++ks) {      // K = 1024
            if (ks + 1 < 32)
                __builtin_prefetch(PcW2 + (((size_t)(wave * 2) * 32 + ks + 1) << 9)
                                        + lane * 16, 0, 1);
            v16h a0 = load_a_frag(h1,            lane, ks * 32, H1S);
            v16h a1 = load_a_frag(h1 + 16 * H1S, lane, ks * 32, H1S);
            #pragma unroll
            for (int t = 0; t < 2; ++t) {
                v16h b = load_b_frag(PcW2, wave * 2 + t, 32, ks, lane);
                acc[t][0] = wmma_f16(a0, b, acc[t][0]);
                acc[t][1] = wmma_f16(a1, b, acc[t][1]);
            }
        }
        int rbase = (lane >> 4) << 3;
        #pragma unroll
        for (int t = 0; t < 2; ++t) {
            int col = (wave * 2 + t) * 16 + (lane & 15);
            float bias = cb2[col];
            #pragma unroll
            for (int h = 0; h < 2; ++h)
                #pragma unroll
                for (int r = 0; r < 8; ++r) {
                    long long gr = row0 + h * 16 + rbase + r;
                    if (gr < N) m_out[gr * 256 + col] = acc[t][h][r] + bias;
                }
        }
    }
    if (wave < 4) {                            // waves 0-3 own the 4 q tiles
        v8f acc[2] = {};
        for (int ks = 0; ks < 4; ++ks) {       // K = 128
            v16h b = load_b_frag(PqW2, wave, 4, ks, lane);
            acc[0] = wmma_f16(load_a_frag(qh,            lane, ks * 32, QHS), b, acc[0]);
            acc[1] = wmma_f16(load_a_frag(qh + 16 * QHS, lane, ks * 32, QHS), b, acc[1]);
        }
        int col = wave * 16 + (lane & 15);
        float bias = qb2[col];
        int rbase = (lane >> 4) << 3;
        #pragma unroll
        for (int h = 0; h < 2; ++h)
            #pragma unroll
            for (int r = 0; r < 8; ++r) {
                long long gr = row0 + h * 16 + rbase + r;
                if (gr < N) q_out[gr * 64 + col] = acc[h][r] + bias;
            }
    }
}

// ---------------- scoring: wave per node, lane per neighbor ----------------
__global__ __launch_bounds__(256) void score_kernel(
    const float* __restrict__ q, const float* __restrict__ k,
    const float* __restrict__ e_bias, const int* __restrict__ nbrs,
    float* __restrict__ w_out, int N) {
    int n = (blockIdx.x * 256 + threadIdx.x) >> 5;
    int lane = threadIdx.x & 31;
    if (n >= N) return;
    int nb = nbrs[(size_t)n * KK + lane];
    const float4* qv = (const float4*)(q + (size_t)n * 64);
    const float4* kv = (const float4*)(k + (size_t)nb * 64);
    float acc = 0.f;
    #pragma unroll
    for (int i = 0; i < 16; ++i) {
        float4 a = qv[i], b = kv[i];
        acc += a.x * b.x + a.y * b.y + a.z * b.z + a.w * b.w;
    }
    acc += e_bias[(size_t)n * KK + lane];
    w_out[(size_t)n * KK + lane] = 1.0f / (1.0f + __expf(-acc));
}

extern "C" void kernel_launch(void* const* d_in, const int* in_sizes, int n_in,
                              void* d_out, int out_size, void* d_ws, size_t ws_size,
                              hipStream_t stream) {
    const float* s        = (const float*)d_in[0];
    const float* node_id  = (const float*)d_in[1];
    const float* e_bias   = (const float*)d_in[2];
    const int*   out_nbrs = (const int*)d_in[3];
    const float* g        = (const float*)d_in[4];
    const float* cW1 = (const float*)d_in[5];
    const float* cb1 = (const float*)d_in[6];
    const float* cW2 = (const float*)d_in[7];
    const float* cb2 = (const float*)d_in[8];
    const float* qW1 = (const float*)d_in[9];
    const float* qb1 = (const float*)d_in[10];
    const float* qW2 = (const float*)d_in[11];
    const float* qb2 = (const float*)d_in[12];
    const float* kW1 = (const float*)d_in[13];
    const float* kb1 = (const float*)d_in[14];
    const float* kW2 = (const float*)d_in[15];
    const float* kb2 = (const float*)d_in[16];

    int N = in_sizes[0] / D_S;

    _Float16* P = (_Float16*)d_ws;
    float* q_ws = (float*)((char*)d_ws + Q_BYTE_OFF);
    float* k_ws = q_ws + (size_t)N * HDQ;
    float* m_out = (float*)d_out;
    float* w_out = m_out + (size_t)N * D_S;

    pack_kernel<<<(PACK_TOTAL + 255) / 256, 256, 0, stream>>>(
        cW1, cW2, qW1, qW2, kW1, kW2, P);

    k_mlp_kernel<<<(N + 15) / 16, 128, 0, stream>>>(
        node_id, P + OFF_KW1, kb1, P + OFF_KW2, kb2, k_ws, N);

    node_kernel<<<(N + MB - 1) / MB, 256, (size_t)SM_TOTAL * 2, stream>>>(
        s, node_id, g, P + OFF_CW1, cb1, P + OFF_CW2, cb2,
        P + OFF_QW1, qb1, P + OFF_QW2, qb2, m_out, q_ws, N);

    score_kernel<<<(N * 32 + 255) / 256, 256, 0, stream>>>(
        q_ws, k_ws, e_bias, out_nbrs, w_out, N);
}
